// SingleHeadAttention_85375359909894
// MI455X (gfx1250) — matrix operand order repaired and checked by reading, not compiled, not run on hardware
//
#include <hip/hip_runtime.h>
#include <hip/hip_bf16.h>

#define B_ 4
#define T_ 4096
#define E_ 1024
#define A_ 128

typedef __attribute__((ext_vector_type(16))) __bf16         v16bf;
typedef __attribute__((ext_vector_type(8)))  float          v8f;
typedef __attribute__((ext_vector_type(8)))  unsigned short v8s;

union BFfrag {
    unsigned short e[16];
    v8s            h[2];
    v16bf          v;
};

__device__ __forceinline__ unsigned short f2bf(float f) {
    union { float f; unsigned u; } x; x.f = f;
    unsigned r = x.u + 0x7FFFu + ((x.u >> 16) & 1u);   // round-to-nearest-even
    return (unsigned short)(r >> 16);
}

__device__ __forceinline__ void cvt8(unsigned short* d, const float4 a, const float4 b) {
    d[0] = f2bf(a.x); d[1] = f2bf(a.y); d[2] = f2bf(a.z); d[3] = f2bf(a.w);
    d[4] = f2bf(b.x); d[5] = f2bf(b.y); d[6] = f2bf(b.z); d[7] = f2bf(b.w);
}

// ---------------------------------------------------------------------------
// One-shot fp32 -> bf16 conversion (used for the weight matrices only).
// ---------------------------------------------------------------------------
__global__ void bf16_convert_kernel(const float* __restrict__ src,
                                    unsigned short* __restrict__ dst, int n8) {
    const int i = blockIdx.x * blockDim.x + threadIdx.x;
    if (i >= n8) return;
    const float4* s = reinterpret_cast<const float4*>(src) + (size_t)i * 2;
    union { unsigned short e[8]; v8s v; } u;
    cvt8(u.e, s[0], s[1]);
    reinterpret_cast<v8s*>(dst)[i] = u.v;
}

// ---------------------------------------------------------------------------
// QKV projection: one wave computes a full 16x128 strip of Y = X @ W^T for
// one of {K,Q,V}. One A-fragment (X, converted fp32->bf16 in-register) feeds
// 8 WMMAs (all N tiles) per k-step; weights are pre-converted bf16 fragment
// loads with no in-loop VALU.
// Outputs bf16: K as [B*T, A], Q (pre-scaled by 1/sqrt(A)) as [B*T, A],
// V transposed as [B, A, T] so PV B-fragments load contiguously.
// ---------------------------------------------------------------------------
__global__ __launch_bounds__(32) void qkv_proj_kernel(
    const float* __restrict__ X,
    const unsigned short* __restrict__ wbf,            // [3][A_][E_] bf16
    unsigned short* __restrict__ kbf, unsigned short* __restrict__ qbf,
    unsigned short* __restrict__ vt)
{
    const int lane = threadIdx.x;
    const int n    = lane & 15;
    const int g    = lane >> 4;
    const int mode = blockIdx.y;                       // 0=K, 1=Q, 2=V
    const unsigned short* W = wbf + (size_t)mode * (A_ * E_);

    const int arow = blockIdx.x * 16 + n;              // A-fragment row for this lane
    const float4* Xr = reinterpret_cast<const float4*>(X + (size_t)arow * E_);

    v8f acc[8];
    #pragma unroll
    for (int j = 0; j < 8; ++j) { v8f z = {}; acc[j] = z; }

    for (int kk = 0; kk < E_; kk += 32) {
        const int kb = kk >> 2;                        // float4 index of kk
        // A-layout: lane row = n; VGPRs 0-3 hold K = g*8..+7, VGPRs 4-7 hold K = 16+g*8..+7
        BFfrag a;
        float4 a0 = Xr[kb + g * 2],     a1 = Xr[kb + g * 2 + 1];
        float4 a2 = Xr[kb + 4 + g * 2], a3 = Xr[kb + 5 + g * 2];
        cvt8(a.e,     a0, a1);
        cvt8(a.e + 8, a2, a3);
        // 8 N tiles share the A fragment
        #pragma unroll
        for (int j = 0; j < 8; ++j) {
            // B-layout: lane col = n; K = g*8..+7 then 16+g*8..+7 (bf16 W row-major in E)
            const v8s* wr = reinterpret_cast<const v8s*>(
                W + (size_t)(j * 16 + n) * E_ + kk);
            BFfrag bfr;
            bfr.h[0] = wr[g];
            bfr.h[1] = wr[2 + g];
            acc[j] = __builtin_amdgcn_wmma_f32_16x16x32_bf16(false, a.v, false, bfr.v,
                                                             (short)0, acc[j], false, false);
        }
    }

    // C-layout: lane col = n, rows = g*8 + r
    #pragma unroll
    for (int r = 0; r < 8; ++r) {
        const int row = blockIdx.x * 16 + g * 8 + r;
        if (mode == 0) {
            #pragma unroll
            for (int j = 0; j < 8; ++j)
                kbf[(size_t)row * A_ + j * 16 + n] = f2bf(acc[j][r]);
        } else if (mode == 1) {
            #pragma unroll
            for (int j = 0; j < 8; ++j)
                qbf[(size_t)row * A_ + j * 16 + n] =
                    f2bf(acc[j][r] * 0.08838834764831845f);  // 1/sqrt(128)
        } else {
            const int bb = row >> 12;                  // T_ = 4096
            const int t  = row & (T_ - 1);
            #pragma unroll
            for (int j = 0; j < 8; ++j)
                vt[((size_t)bb * A_ + j * 16 + n) * T_ + t] = f2bf(acc[j][r]);
        }
    }
}

// ---------------------------------------------------------------------------
// Flash attention: one wave per (batch, 16 query rows). Online softmax with
// 64-key steps: 16 WMMA for scores + 16 WMMA for PV per step, so shuffle
// reductions and accumulator rescales amortize over twice as many keys.
// ---------------------------------------------------------------------------
__global__ __launch_bounds__(32) void attn_kernel(
    const unsigned short* __restrict__ qbf,
    const unsigned short* __restrict__ kbf,
    const unsigned short* __restrict__ vt,
    float* __restrict__ out)
{
    __shared__ __align__(16) unsigned short Pl[16 * 64]; // P tile staging (C->A transpose)

    const int lane = threadIdx.x;
    const int n    = lane & 15;
    const int g    = lane >> 4;
    const int qt   = blockIdx.x;
    const int b    = blockIdx.y;
    const int qm0  = qt * 16;

    // Persistent Q fragments: 16x128 bf16 as 4 A-fragments (32 VGPRs)
    BFfrag qf[4];
    {
        const v8s* qp = reinterpret_cast<const v8s*>(qbf + ((size_t)(b * T_) + qm0 + n) * A_);
        #pragma unroll
        for (int ka = 0; ka < 4; ++ka) {
            qf[ka].h[0] = qp[ka * 4 + g];
            qf[ka].h[1] = qp[ka * 4 + 2 + g];
        }
    }

    float mrow[8], lrow[8];
    v8f   acc[8];
    #pragma unroll
    for (int r = 0; r < 8; ++r) { mrow[r] = -1e30f; lrow[r] = 0.0f; }
    #pragma unroll
    for (int j = 0; j < 8; ++j) { v8f z = {}; acc[j] = z; }

    const int nsteps = qt / 4 + 1;                     // 64-key steps covering [0, qm0+16)
    for (int step = 0; step < nsteps; ++step) {
        const int s0 = step * 64;

        if (step + 1 < nsteps) {                       // pull next K/V tiles toward L2/L0
            __builtin_prefetch(kbf + ((size_t)(b * T_) + s0 + 64 + n) * A_, 0, 1);
            __builtin_prefetch(vt + ((size_t)b * A_ + n) * T_ + s0 + 64, 0, 1);
        }

        // ----- scores: four 16x16 tiles (cols s0+h*16 .. s0+h*16+15) -----
        v8f cs[4];
        #pragma unroll
        for (int h = 0; h < 4; ++h) {
            const v8s* kp = reinterpret_cast<const v8s*>(
                kbf + ((size_t)(b * T_) + s0 + h * 16 + n) * A_);
            v8f cc = {};
            #pragma unroll
            for (int ka = 0; ka < 4; ++ka) {
                BFfrag kf;                             // B-layout: lane col = key n, K = g*8..+7, 16+g*8..+7
                kf.h[0] = kp[ka * 4 + g];
                kf.h[1] = kp[ka * 4 + 2 + g];
                cc = __builtin_amdgcn_wmma_f32_16x16x32_bf16(false, qf[ka].v, false, kf.v,
                                                             (short)0, cc, false, false);
            }
            cs[h] = cc;
        }

        // ----- online softmax + write P (bf16) to LDS -----
        const bool domask = (s0 + 63 > qm0);           // only the diagonal step needs masking
        #pragma unroll
        for (int r = 0; r < 8; ++r) {
            const int qrow = qm0 + g * 8 + r;
            float x[4];
            #pragma unroll
            for (int h = 0; h < 4; ++h) {
                x[h] = cs[h][r];
                if (domask && (s0 + h * 16 + n > qrow)) x[h] = -1e30f;
            }
            float t = fmaxf(fmaxf(x[0], x[1]), fmaxf(x[2], x[3]));
            t = fmaxf(t, __shfl_xor(t, 1));            // row reduce across the 16-lane half
            t = fmaxf(t, __shfl_xor(t, 2));
            t = fmaxf(t, __shfl_xor(t, 4));
            t = fmaxf(t, __shfl_xor(t, 8));
            const float mnew  = fmaxf(mrow[r], t);
            const float alpha = __expf(mrow[r] - mnew);
            float p[4];
            #pragma unroll
            for (int h = 0; h < 4; ++h) p[h] = __expf(x[h] - mnew);
            float rs = (p[0] + p[1]) + (p[2] + p[3]);
            rs += __shfl_xor(rs, 1);
            rs += __shfl_xor(rs, 2);
            rs += __shfl_xor(rs, 4);
            rs += __shfl_xor(rs, 8);
            lrow[r] = lrow[r] * alpha + rs;
            mrow[r] = mnew;
            #pragma unroll
            for (int j = 0; j < 8; ++j) acc[j][r] *= alpha;
            #pragma unroll
            for (int h = 0; h < 4; ++h)
                Pl[(g * 8 + r) * 64 + h * 16 + n] = f2bf(p[h]);
        }
        __syncthreads();                               // single-wave WG: near-free

        // ----- P as two A-fragments (16x32 each) from LDS -----
        BFfrag pa[2];
        {
            const v8s* pp = reinterpret_cast<const v8s*>(Pl + n * 64);
            pa[0].h[0] = pp[g * 2];     pa[0].h[1] = pp[g * 2 + 1];   // cols 0..31
            pa[1].h[0] = pp[4 + g * 2]; pa[1].h[1] = pp[5 + g * 2];   // cols 32..63
        }

        // ----- O += P @ V over 8 column tiles x 2 key chunks -----
        #pragma unroll
        for (int j = 0; j < 8; ++j) {
            const int a = j * 16 + n;
            const v8s* vp = reinterpret_cast<const v8s*>(
                vt + ((size_t)b * A_ + a) * T_ + s0);  // V^T: s contiguous per column a
            BFfrag vf0, vf1;
            vf0.h[0] = vp[g * 2];     vf0.h[1] = vp[g * 2 + 1];      // s = s0 .. s0+31
            vf1.h[0] = vp[4 + g * 2]; vf1.h[1] = vp[5 + g * 2];      // s = s0+32 .. s0+63
            acc[j] = __builtin_amdgcn_wmma_f32_16x16x32_bf16(false, pa[0].v, false, vf0.v,
                                                             (short)0, acc[j], false, false);
            acc[j] = __builtin_amdgcn_wmma_f32_16x16x32_bf16(false, pa[1].v, false, vf1.v,
                                                             (short)0, acc[j], false, false);
        }
        __syncthreads();
    }

    // ----- finalize: O / l, fp32 out [B,T,A] -----
    #pragma unroll
    for (int r = 0; r < 8; ++r) {
        const float inv = 1.0f / lrow[r];
        const int trow  = qm0 + g * 8 + r;
        float* orow = out + ((size_t)(b * T_) + trow) * A_;
        #pragma unroll
        for (int j = 0; j < 8; ++j) orow[j * 16 + n] = acc[j][r] * inv;
    }
}

extern "C" void kernel_launch(void* const* d_in, const int* in_sizes, int n_in,
                              void* d_out, int out_size, void* d_ws, size_t ws_size,
                              hipStream_t stream) {
    (void)in_sizes; (void)n_in; (void)out_size;
    const float* X  = (const float*)d_in[0];
    const float* Wk = (const float*)d_in[1];
    const float* Wq = (const float*)d_in[2];
    const float* Wv = (const float*)d_in[3];
    float* out = (float*)d_out;

    const size_t nproj = (size_t)B_ * T_ * A_;         // 2,097,152 elements each
    const size_t nw    = (size_t)A_ * E_;              // 131,072 elements per weight
    const size_t need  = (3 * nproj + 3 * nw) * sizeof(unsigned short); // ~12.75 MB
    if (ws_size < need) return;

    unsigned short* kbf = (unsigned short*)d_ws;
    unsigned short* qbf = kbf + nproj;
    unsigned short* vt  = qbf + nproj;
    unsigned short* wbf = vt + nproj;                  // [3][A_][E_]

    // 1) Weights fp32 -> bf16 (once, 0.75 MB total)
    const int n8 = (int)(nw / 8);
    bf16_convert_kernel<<<dim3((n8 + 255) / 256), dim3(256), 0, stream>>>(Wk, wbf, n8);
    bf16_convert_kernel<<<dim3((n8 + 255) / 256), dim3(256), 0, stream>>>(Wq, wbf + nw, n8);
    bf16_convert_kernel<<<dim3((n8 + 255) / 256), dim3(256), 0, stream>>>(Wv, wbf + 2 * nw, n8);

    // 2) Projections: grid (M tiles, {K,Q,V}); each wave = 16x128 strip, 8 WMMA per k-step
    qkv_proj_kernel<<<dim3((B_ * T_) / 16, 3), dim3(32), 0, stream>>>(X, wbf, kbf, qbf, vt);

    // 3) Flash attention: 64-key steps
    attn_kernel<<<dim3(T_ / 16, B_), dim3(32), 0, stream>>>(qbf, kbf, vt, out);
}
